// SelfAttention_71167608095234
// MI455X (gfx1250) — matrix operand order repaired and checked by reading, not compiled, not run on hardware
//
#include <hip/hip_runtime.h>
#include <hip/hip_bf16.h>

typedef _Float16 half_t;
typedef __attribute__((ext_vector_type(16))) _Float16 v16h;
typedef __attribute__((ext_vector_type(8)))  _Float16 v8h;
typedef __attribute__((ext_vector_type(4)))  _Float16 v4h;
typedef __attribute__((ext_vector_type(8)))  float    v8f;
typedef __attribute__((ext_vector_type(4)))  int      v4i;

union V16 { v16h v; v8h h[2]; };

#define WMMA_F16(a, b, c) \
  __builtin_amdgcn_wmma_f32_16x16x32_f16(false, (a), false, (b), (short)0, (c), false, false)

#if __has_builtin(__builtin_amdgcn_global_load_async_to_lds_b128)
#define HAS_ASYNC_CP 1
#endif
#if __has_builtin(__builtin_amdgcn_s_wait_asynccnt)
#define ASYNC_WAIT() __builtin_amdgcn_s_wait_asynccnt(0)
#else
#define ASYNC_WAIT()
#endif

constexpr int B  = 4;
constexpr int N  = 2048;
constexpr int C  = 512;
constexpr int H  = 8;
constexpr int D  = 64;
constexpr int BN = B * N;     // 8192
constexpr int C3 = 3 * C;     // 1536

// LDS pitches (halves) chosen so 16 consecutive rows hit distinct banks
constexpr int KP = 72;        // K stage: 32 rows x 64 halves, pitch 72 (144B)
constexpr int VP = 40;        // V stage: 64 rows x 32 halves, pitch 40 (80B)
constexpr int PP = 40;        // P stage: 16 rows x 32 halves, pitch 40

// 16B global -> LDS copy: async DMA when available, sync fallback otherwise
__device__ inline void cp16(half_t* l, const half_t* g) {
#ifdef HAS_ASYNC_CP
  __builtin_amdgcn_global_load_async_to_lds_b128(
      (__attribute__((address_space(1))) v4i*)g,
      (__attribute__((address_space(3))) v4i*)l, 0, 0);
#else
  *(v8h*)l = *(const v8h*)g;
#endif
}

// ---------------------------------------------------------------------------
// Kernel 1a: fp32 -> f16 copy (vectorized x4)
// ---------------------------------------------------------------------------
__global__ void cvt_f32_f16(const float* __restrict__ in, half_t* __restrict__ out, int n) {
  int i = (blockIdx.x * blockDim.x + threadIdx.x) * 4;
  if (i < n) {
    float4 f = *(const float4*)(in + i);
    v4h h;
    h[0] = (half_t)f.x; h[1] = (half_t)f.y; h[2] = (half_t)f.z; h[3] = (half_t)f.w;
    *(v4h*)(out + i) = h;
  }
}

// ---------------------------------------------------------------------------
// Kernel 1b: fp32 [K][M] -> f16 [M][K] transpose-convert (weights)
// ---------------------------------------------------------------------------
__global__ void transpose_cvt(const float* __restrict__ in, half_t* __restrict__ out,
                              int K, int M) {
  int i = blockIdx.x * blockDim.x + threadIdx.x;
  if (i < K * M) {
    int k = i / M, m = i % M;
    out[(size_t)m * K + k] = (half_t)in[i];
  }
}

// ---------------------------------------------------------------------------
// QKV tile scatter: one 16x16 fp32 C-tile -> Q[B,H,N,D] / K[B,H,N,D] / Vt[B,H,D,N]
// ---------------------------------------------------------------------------
__device__ inline void store_qkv_tile(const v8f& acc, int m0, int c0, int lo, int hi,
                                      half_t* __restrict__ Qh, half_t* __restrict__ Kh,
                                      half_t* __restrict__ Vt) {
  const int which = c0 / C;                  // 0=Q 1=K 2=V
  const int rem   = c0 % C;
  const int h     = rem / D;
  const int d0    = rem % D;
  const int b_    = m0 / N;
  const int n0    = m0 % N;
  const int bh    = b_ * H + h;
  const int d     = d0 + lo;
  if (which == 2) {
    v8h t;
#pragma unroll
    for (int r = 0; r < 8; ++r) t[r] = (half_t)acc[r];
    *(v8h*)(Vt + ((size_t)bh * D + d) * N + n0 + hi * 8) = t;
  } else {
    half_t* dst = (which == 0) ? Qh : Kh;
#pragma unroll
    for (int r = 0; r < 8; ++r) {
      int n = n0 + r + hi * 8;
      dst[((size_t)bh * N + n) * D + d] = (half_t)acc[r];
    }
  }
}

// ---------------------------------------------------------------------------
// Kernel 2: QKV GEMM, 32x32 output tile per wave (2x2 register blocking).
// ---------------------------------------------------------------------------
__global__ __launch_bounds__(256) void qkv_gemm(const half_t* __restrict__ Xh,
                                                const half_t* __restrict__ Wt,
                                                half_t* __restrict__ Qh,
                                                half_t* __restrict__ Kh,
                                                half_t* __restrict__ Vt) {
  const int lane = threadIdx.x & 31;
  const int wave = threadIdx.x >> 5;
  const int tile = blockIdx.x * 8 + wave;
  const int tcols = C3 / 32;                 // 48
  const int m0 = (tile / tcols) * 32;
  const int c0 = (tile % tcols) * 32;
  const int hi = lane >> 4;
  const int lo = lane & 15;

  v8f a00 = {}, a01 = {}, a10 = {}, a11 = {};
  const half_t* ar0 = Xh + (size_t)(m0 + lo) * C;
  const half_t* ar1 = Xh + (size_t)(m0 + 16 + lo) * C;
  const half_t* br0 = Wt + (size_t)(c0 + lo) * C;
  const half_t* br1 = Wt + (size_t)(c0 + 16 + lo) * C;
#pragma unroll
  for (int k0 = 0; k0 < C; k0 += 32) {
    V16 fa0, fa1, fb0, fb1;
    fa0.h[0] = *(const v8h*)(ar0 + k0 + hi * 16);
    fa0.h[1] = *(const v8h*)(ar0 + k0 + hi * 16 + 8);
    fa1.h[0] = *(const v8h*)(ar1 + k0 + hi * 16);
    fa1.h[1] = *(const v8h*)(ar1 + k0 + hi * 16 + 8);
    fb0.h[0] = *(const v8h*)(br0 + k0 + hi * 16);
    fb0.h[1] = *(const v8h*)(br0 + k0 + hi * 16 + 8);
    fb1.h[0] = *(const v8h*)(br1 + k0 + hi * 16);
    fb1.h[1] = *(const v8h*)(br1 + k0 + hi * 16 + 8);
    a00 = WMMA_F16(fa0.v, fb0.v, a00);
    a01 = WMMA_F16(fa0.v, fb1.v, a01);
    a10 = WMMA_F16(fa1.v, fb0.v, a10);
    a11 = WMMA_F16(fa1.v, fb1.v, a11);
  }
  store_qkv_tile(a00, m0,      c0,      lo, hi, Qh, Kh, Vt);
  store_qkv_tile(a01, m0,      c0 + 16, lo, hi, Qh, Kh, Vt);
  store_qkv_tile(a10, m0 + 16, c0,      lo, hi, Qh, Kh, Vt);
  store_qkv_tile(a11, m0 + 16, c0 + 16, lo, hi, Qh, Kh, Vt);
}

// ---------------------------------------------------------------------------
// Kernel 3: flash attention. Grid = B*H*(N/128) blocks, 8 waves/block.
// K/V blocks staged in LDS (async copy, double buffered, shared by 8 waves).
// Each wave: 16 query rows x D=64, online softmax over 32-key blocks.
// ---------------------------------------------------------------------------
__global__ __launch_bounds__(256) void flash_attn(const half_t* __restrict__ Qh,
                                                  const half_t* __restrict__ Kh,
                                                  const half_t* __restrict__ Vt,
                                                  half_t* __restrict__ AO) {
  __shared__ half_t sK[2][32 * KP];          // 2 x 4.5 KB
  __shared__ half_t sV[2][64 * VP];          // 2 x 5 KB
  __shared__ half_t sP[8][16 * PP];          // 10 KB (per-wave P staging)

  const int lane = threadIdx.x & 31;
  const int wave = threadIdx.x >> 5;
  const int hi = lane >> 4, lo = lane & 15;
  const int bh   = blockIdx.x >> 4;          // N/128 == 16
  const int qblk = blockIdx.x & 15;
  const int q0 = qblk * 128 + wave * 16;

  const half_t* Kg = Kh + (size_t)bh * N * D;
  const half_t* Vg = Vt + (size_t)bh * D * N;

  // cooperative copy assignment: 256 threads x 16B cover one 4KB block each
  const int ck = wave * 32 + lane;           // 0..255
  const half_t* gK = Kg + ck * 8;            // K block is contiguous 4KB
  const int ldsK = (ck >> 3) * KP + (ck & 7) * 8;
  const int vrow = ck >> 2, vch = ck & 3;    // V block: 64 rows x 64B
  const half_t* gV = Vg + (size_t)vrow * N + vch * 8;
  const int ldsV = vrow * VP + vch * 8;

  // prefetch first K/V block into buffer 0
  cp16(&sK[0][ldsK], gK);
  cp16(&sV[0][ldsV], gV);

  // Q A-fragments (global, reused across all iterations)
  const half_t* Qrow = Qh + ((size_t)bh * N + q0 + lo) * D;
  V16 aq0, aq1;
  aq0.h[0] = *(const v8h*)(Qrow + hi * 16);
  aq0.h[1] = *(const v8h*)(Qrow + hi * 16 + 8);
  aq1.h[0] = *(const v8h*)(Qrow + 32 + hi * 16);
  aq1.h[1] = *(const v8h*)(Qrow + 32 + hi * 16 + 8);

  float m[8], l[8], alpha[8];
#pragma unroll
  for (int r = 0; r < 8; ++r) { m[r] = -1e30f; l[r] = 0.f; }
  v8f O0 = {}, O1 = {}, O2 = {}, O3 = {};
  const float scale = 0.125f;                // D^-0.5

  ASYNC_WAIT();
  __syncthreads();

  for (int j0 = 0; j0 < N; j0 += 32) {
    const int cur = (j0 >> 5) & 1;
    if (j0 + 32 < N) {                       // prefetch next block
      cp16(&sK[cur ^ 1][ldsK], gK + (j0 + 32) * D);
      cp16(&sV[cur ^ 1][ldsV], gV + (j0 + 32));
    }
    const half_t* KB = &sK[cur][0];
    const half_t* VB = &sV[cur][0];

    // ---- S = Q @ K^T for two 16-key column tiles (B-frags from LDS)
    v8f S0 = {}, S1 = {};
    {
      const half_t* kr0 = KB + lo * KP;
      const half_t* kr1 = KB + (16 + lo) * KP;
      V16 b0, b1;
      b0.h[0] = *(const v8h*)(kr0 + hi * 16);
      b0.h[1] = *(const v8h*)(kr0 + hi * 16 + 8);
      b1.h[0] = *(const v8h*)(kr0 + 32 + hi * 16);
      b1.h[1] = *(const v8h*)(kr0 + 32 + hi * 16 + 8);
      S0 = WMMA_F16(aq0.v, b0.v, S0);
      S0 = WMMA_F16(aq1.v, b1.v, S0);
      b0.h[0] = *(const v8h*)(kr1 + hi * 16);
      b0.h[1] = *(const v8h*)(kr1 + hi * 16 + 8);
      b1.h[0] = *(const v8h*)(kr1 + 32 + hi * 16);
      b1.h[1] = *(const v8h*)(kr1 + 32 + hi * 16 + 8);
      S1 = WMMA_F16(aq0.v, b0.v, S1);
      S1 = WMMA_F16(aq1.v, b1.v, S1);
    }

    // ---- online softmax (lanes 0-15 hold M=r, lanes 16-31 hold M=r+8)
#pragma unroll
    for (int r = 0; r < 8; ++r) {
      float s0 = S0[r] * scale, s1 = S1[r] * scale;
      float rm = fmaxf(s0, s1);
      rm = fmaxf(rm, __shfl_xor(rm, 1, 32));
      rm = fmaxf(rm, __shfl_xor(rm, 2, 32));
      rm = fmaxf(rm, __shfl_xor(rm, 4, 32));
      rm = fmaxf(rm, __shfl_xor(rm, 8, 32));
      float nm = fmaxf(m[r], rm);
      alpha[r] = __expf(m[r] - nm);
      float p0 = __expf(s0 - nm);
      float p1 = __expf(s1 - nm);
      float rs = p0 + p1;
      rs += __shfl_xor(rs, 1, 32);
      rs += __shfl_xor(rs, 2, 32);
      rs += __shfl_xor(rs, 4, 32);
      rs += __shfl_xor(rs, 8, 32);
      l[r] = l[r] * alpha[r] + rs;
      m[r] = nm;
      S0[r] = p0; S1[r] = p1;
    }
#pragma unroll
    for (int r = 0; r < 8; ++r) {
      O0[r] *= alpha[r]; O1[r] *= alpha[r];
      O2[r] *= alpha[r]; O3[r] *= alpha[r];
    }

    // ---- P: C-layout -> A-layout via per-wave LDS tile (same-wave, no barrier)
    half_t* Pw = &sP[wave][0];
#pragma unroll
    for (int r = 0; r < 8; ++r) {
      int row = r + hi * 8;
      Pw[row * PP + lo]      = (half_t)S0[r];
      Pw[row * PP + 16 + lo] = (half_t)S1[r];
    }
    V16 pa;
    pa.h[0] = *(const v8h*)(Pw + lo * PP + hi * 16);
    pa.h[1] = *(const v8h*)(Pw + lo * PP + hi * 16 + 8);

    // ---- O += P @ V  (V B-frags from LDS, transposed layout)
    V16 bv;
    const half_t* v0 = VB + lo * VP + hi * 16;
    bv.h[0] = *(const v8h*)(v0);                 bv.h[1] = *(const v8h*)(v0 + 8);
    O0 = WMMA_F16(pa.v, bv.v, O0);
    bv.h[0] = *(const v8h*)(v0 + 16 * VP);       bv.h[1] = *(const v8h*)(v0 + 16 * VP + 8);
    O1 = WMMA_F16(pa.v, bv.v, O1);
    bv.h[0] = *(const v8h*)(v0 + 32 * VP);       bv.h[1] = *(const v8h*)(v0 + 32 * VP + 8);
    O2 = WMMA_F16(pa.v, bv.v, O2);
    bv.h[0] = *(const v8h*)(v0 + 48 * VP);       bv.h[1] = *(const v8h*)(v0 + 48 * VP + 8);
    O3 = WMMA_F16(pa.v, bv.v, O3);

    ASYNC_WAIT();                               // this wave's prefetch done
    __syncthreads();                            // all waves done reading cur
  }

  // ---- epilogue: normalize, store f16 [B, N, H*D]
  const int b_idx = bh / H, h_idx = bh % H;
#pragma unroll
  for (int r = 0; r < 8; ++r) {
    int n = q0 + r + hi * 8;
    float inv = 1.0f / l[r];
    size_t base = ((size_t)b_idx * N + n) * C + h_idx * D;
    AO[base +  0 + lo] = (half_t)(O0[r] * inv);
    AO[base + 16 + lo] = (half_t)(O1[r] * inv);
    AO[base + 32 + lo] = (half_t)(O2[r] * inv);
    AO[base + 48 + lo] = (half_t)(O3[r] * inv);
  }
}

// ---------------------------------------------------------------------------
// Kernel 4: out = AO[8192x512] @ W_out + bias, 32x32 tile per wave, fp32 out
// ---------------------------------------------------------------------------
__global__ __launch_bounds__(256) void out_gemm(const half_t* __restrict__ AO,
                                                const half_t* __restrict__ WtO,
                                                const float* __restrict__ bias,
                                                float* __restrict__ out) {
  const int lane = threadIdx.x & 31;
  const int wave = threadIdx.x >> 5;
  const int tile = blockIdx.x * 8 + wave;
  const int tcols = C / 32;                  // 16
  const int m0 = (tile / tcols) * 32;
  const int c0 = (tile % tcols) * 32;
  const int hi = lane >> 4, lo = lane & 15;

  v8f a00 = {}, a01 = {}, a10 = {}, a11 = {};
  const half_t* ar0 = AO  + (size_t)(m0 + lo) * C;
  const half_t* ar1 = AO  + (size_t)(m0 + 16 + lo) * C;
  const half_t* br0 = WtO + (size_t)(c0 + lo) * C;
  const half_t* br1 = WtO + (size_t)(c0 + 16 + lo) * C;
#pragma unroll
  for (int k0 = 0; k0 < C; k0 += 32) {
    V16 fa0, fa1, fb0, fb1;
    fa0.h[0] = *(const v8h*)(ar0 + k0 + hi * 16);
    fa0.h[1] = *(const v8h*)(ar0 + k0 + hi * 16 + 8);
    fa1.h[0] = *(const v8h*)(ar1 + k0 + hi * 16);
    fa1.h[1] = *(const v8h*)(ar1 + k0 + hi * 16 + 8);
    fb0.h[0] = *(const v8h*)(br0 + k0 + hi * 16);
    fb0.h[1] = *(const v8h*)(br0 + k0 + hi * 16 + 8);
    fb1.h[0] = *(const v8h*)(br1 + k0 + hi * 16);
    fb1.h[1] = *(const v8h*)(br1 + k0 + hi * 16 + 8);
    a00 = WMMA_F16(fa0.v, fb0.v, a00);
    a01 = WMMA_F16(fa0.v, fb1.v, a01);
    a10 = WMMA_F16(fa1.v, fb0.v, a10);
    a11 = WMMA_F16(fa1.v, fb1.v, a11);
  }
  float bb0 = bias[c0 + lo];
  float bb1 = bias[c0 + 16 + lo];
#pragma unroll
  for (int r = 0; r < 8; ++r) {
    int mr0 = m0 + r + hi * 8;
    int mr1 = m0 + 16 + r + hi * 8;
    out[(size_t)mr0 * C + c0 + lo]      = a00[r] + bb0;
    out[(size_t)mr0 * C + c0 + 16 + lo] = a01[r] + bb1;
    out[(size_t)mr1 * C + c0 + lo]      = a10[r] + bb0;
    out[(size_t)mr1 * C + c0 + 16 + lo] = a11[r] + bb1;
  }
}

// ---------------------------------------------------------------------------
extern "C" void kernel_launch(void* const* d_in, const int* in_sizes, int n_in,
                              void* d_out, int out_size, void* d_ws, size_t ws_size,
                              hipStream_t stream) {
  const float* x     = (const float*)d_in[0];
  const float* W_in  = (const float*)d_in[1];
  const float* W_out = (const float*)d_in[2];
  const float* b_out = (const float*)d_in[3];
  float* out = (float*)d_out;

  char* ws = (char*)d_ws;
  half_t* Xh    = (half_t*)ws; ws += (size_t)BN * C * 2;        // 8 MB
  half_t* WtIn  = (half_t*)ws; ws += (size_t)C3 * C * 2;        // 1.5 MB
  half_t* WtOut = (half_t*)ws; ws += (size_t)C * C * 2;         // 0.5 MB
  half_t* Qh    = (half_t*)ws; ws += (size_t)B * H * N * D * 2; // 8 MB
  half_t* Kh    = (half_t*)ws; ws += (size_t)B * H * N * D * 2; // 8 MB
  half_t* Vt    = (half_t*)ws; ws += (size_t)B * H * N * D * 2; // 8 MB
  half_t* AO    = (half_t*)ws;                                  // 8 MB

  cvt_f32_f16<<<(BN * C / 4 + 255) / 256, 256, 0, stream>>>(x, Xh, BN * C);
  transpose_cvt<<<(C * C3 + 255) / 256, 256, 0, stream>>>(W_in, WtIn, C, C3);
  transpose_cvt<<<(C * C + 255) / 256, 256, 0, stream>>>(W_out, WtOut, C, C);

  qkv_gemm<<<(BN / 32) * (C3 / 32) / 8, 256, 0, stream>>>(Xh, WtIn, Qh, Kh, Vt);
  flash_attn<<<B * H * (N / 128), 256, 0, stream>>>(Qh, Kh, Vt, AO);
  out_gemm<<<(BN / 32) * (C / 32) / 8, 256, 0, stream>>>(AO, WtOut, b_out, out);
}
